// GAT_5437428597510
// MI455X (gfx1250) — matrix operand order repaired and checked by reading, not compiled, not run on hardware
//
#include <hip/hip_runtime.h>
#include <hip/hip_bf16.h>
#include <math.h>

#define NNODES 50000
#define NEDGES 800000
#define EO     850000   /* edges + self loops */
#define FIN    256
#define HC     256      /* heads * dim_h */
#define NHEADS 4
#define CH     64
#define C2     10
#define NEG    0.2f

typedef __attribute__((ext_vector_type(16))) __bf16 v16bf;
typedef __attribute__((ext_vector_type(8)))  float  v8f;

union V16 { v16bf v; unsigned short u[16]; uint4 q[2]; };
union PK8 { unsigned short s[8]; uint4 q; };

__device__ __forceinline__ unsigned short f2bf(float f) {
    unsigned u = __float_as_uint(f);
    u += 0x7FFFu + ((u >> 16) & 1u);          // round-to-nearest-even
    return (unsigned short)(u >> 16);
}
__device__ __forceinline__ unsigned fkey(float f) {   // order-preserving float->u32
    unsigned u = __float_as_uint(f);
    return u ^ ((unsigned)((int)u >> 31) | 0x80000000u);
}
__device__ __forceinline__ float fdec(unsigned k) {
    unsigned u = (k & 0x80000000u) ? (k ^ 0x80000000u) : ~k;
    return __uint_as_float(u);
}
__device__ __forceinline__ float lrelu(float v) { return v > 0.f ? v : NEG * v; }

// ---------------- utility: zero fill ----------------
__global__ void fill_zero_kernel(unsigned* p, long long n) {
    long long i = (long long)blockIdx.x * blockDim.x + threadIdx.x;
    if (i < n) p[i] = 0u;
}

// ---------------- fp32 -> packed bf16 convert (8 elems/thread) --------------
__global__ void cvt_bf16_kernel(const float* __restrict__ src,
                                unsigned short* __restrict__ dst, long long n8) {
    long long t = (long long)blockIdx.x * blockDim.x + threadIdx.x;
    if (t >= n8) return;
    const float4* p = (const float4*)(src + t * 8);
    float4 a = p[0], b = p[1];
    PK8 o;
    o.s[0] = f2bf(a.x); o.s[1] = f2bf(a.y); o.s[2] = f2bf(a.z); o.s[3] = f2bf(a.w);
    o.s[4] = f2bf(b.x); o.s[5] = f2bf(b.y); o.s[6] = f2bf(b.z); o.s[7] = f2bf(b.w);
    *(uint4*)(dst + t * 8) = o.q;
}

// ---------------- GEMM: C[M x ncols] = A[M x 256](bf16) * B[256 x ncols](f32)
// bf16 WMMA, f32 accumulate. One wave per 16x16 tile; 4 waves/block share a
// 256x16 B tile staged in LDS *pre-swizzled into per-lane fragment layout*:
// BsF[kstep][lane][j] so each lane loads its fragment as 2x ds_load_b128.
__global__ void gemm_wmma_bf16(const unsigned short* __restrict__ A,
                               const float* __restrict__ B,
                               float* __restrict__ C,
                               int Mtiles, int ncols, int Ntiles) {
    __shared__ unsigned short BsF[8 * 32 * 16];   // 8 ksteps x 32 lanes x 16 bf16
    const int tn = blockIdx.x % Ntiles;
    const int tmBase = (blockIdx.x / Ntiles) * 4;

    // stage: chunk c covers 8 consecutive fragment elements (one ds_store_b128)
    for (int c = threadIdx.x; c < 512; c += blockDim.x) {
        int j0     = (c & 1) * 8;
        int lane_s = (c >> 1) & 31;
        int kstep  = c >> 6;
        int n      = lane_s & 15;
        int col    = tn * 16 + n;
        int kbase  = kstep * 32 + (lane_s >> 4) * 8 + j0 * 2;
        PK8 o;
        if (col < ncols) {
            #pragma unroll
            for (int i = 0; i < 8; ++i)
                o.s[i] = f2bf(B[(long long)(kbase + i) * ncols + col]);
        } else {
            #pragma unroll
            for (int i = 0; i < 8; ++i) o.s[i] = 0;
        }
        *(uint4*)(BsF + c * 8) = o.q;
    }
    __syncthreads();

    const int wave = threadIdx.x >> 5;
    const int lane = threadIdx.x & 31;
    const int tm = tmBase + wave;
    if (tm >= Mtiles) return;                 // wave-uniform: EXEC stays all-1s
    const int mrow = lane & 15;
    const int half = lane >> 4;
    const unsigned short* rowA = A + (long long)(tm * 16 + mrow) * 256;

    v8f c = {};
    #pragma unroll
    for (int ks = 0; ks < 8; ++ks) {
        V16 a, b;
        // A 16x32 fragment: per lane two contiguous 8-elem bf16 runs
        //   run0: kb + half*8 .. +7 ; run1: kb + 16 + half*8 .. +7
        const uint4* pa = (const uint4*)(rowA + ks * 32 + half * 8);
        a.q[0] = pa[0];
        a.q[1] = pa[2];
        // B 32x16 fragment: pre-swizzled, contiguous 32B per lane
        const uint4* pb = (const uint4*)(BsF + ks * 512 + lane * 16);
        b.q[0] = pb[0];
        b.q[1] = pb[1];
        c = __builtin_amdgcn_wmma_f32_16x16x32_bf16(false, a.v, false, b.v,
                                                    (short)0, c, false, false);
    }
    const int col = tn * 16 + mrow;
    if (col < ncols) {
        #pragma unroll
        for (int v = 0; v < 8; ++v) {         // C/D: VGPR v -> M = v + 8*half
            int r = tm * 16 + v + half * 8;
            C[(long long)r * ncols + col] = c[v];
        }
    }
}

// ---------------- layer 1 edge kernels (wave per edge) ----------------------
__global__ void score1_kernel(const int* __restrict__ ei,
                              const float* __restrict__ xl,
                              const float* __restrict__ xr,
                              const float* __restrict__ att,
                              float* __restrict__ sc,
                              unsigned* __restrict__ mkey) {
    int e = blockIdx.x * 8 + (threadIdx.x >> 5);
    if (e >= EO) return;
    int lane = threadIdx.x & 31;
    int h = lane >> 3, sub = lane & 7;
    int s, d;
    if (e < NEDGES) { s = ei[e]; d = ei[NEDGES + e]; } else { s = d = e - NEDGES; }
    const float* ps = xl + (long long)s * HC + h * CH + sub * 8;
    const float* pd = xr + (long long)d * HC + h * CH + sub * 8;
    const float* pa = att + h * CH + sub * 8;
    float acc = 0.f;
    #pragma unroll
    for (int i = 0; i < 8; ++i) acc += pa[i] * lrelu(ps[i] + pd[i]);
    acc += __shfl_xor(acc, 1, 32);
    acc += __shfl_xor(acc, 2, 32);
    acc += __shfl_xor(acc, 4, 32);
    if (sub == 0) {
        sc[(long long)e * NHEADS + h] = acc;
        atomicMax(&mkey[d * NHEADS + h], fkey(acc));
    }
}

__global__ void ex1_kernel(const int* __restrict__ ei, float* __restrict__ sc,
                           const unsigned* __restrict__ mkey, float* __restrict__ den) {
    long long t = (long long)blockIdx.x * blockDim.x + threadIdx.x;
    if (t >= (long long)EO * NHEADS) return;
    int e = (int)(t >> 2), h = (int)(t & 3);
    int d = (e < NEDGES) ? ei[NEDGES + e] : e - NEDGES;
    float ex = __expf(sc[t] - fdec(mkey[d * NHEADS + h]));
    sc[t] = ex;
    atomicAdd(&den[d * NHEADS + h], ex);
}

__global__ void agg1_kernel(const int* __restrict__ ei, const float* __restrict__ xl,
                            const float* __restrict__ sc, const float* __restrict__ den,
                            float* __restrict__ hout) {
    int e = blockIdx.x * 8 + (threadIdx.x >> 5);
    if (e >= EO) return;
    int lane = threadIdx.x & 31;
    int h = lane >> 3, sub = lane & 7;
    int s, d;
    if (e < NEDGES) { s = ei[e]; d = ei[NEDGES + e]; } else { s = d = e - NEDGES; }
    float alpha = sc[(long long)e * NHEADS + h] / den[d * NHEADS + h];
    const float* ps = xl + (long long)s * HC + h * CH + sub * 8;
    float* pd = hout + (long long)d * HC + h * CH + sub * 8;
    #pragma unroll
    for (int i = 0; i < 8; ++i) atomicAdd(&pd[i], ps[i] * alpha);
}

// bias + relu, writing both not needed: only bf16 copy is consumed downstream
__global__ void biasrelu_bf16_kernel(const float* __restrict__ h,
                                     const float* __restrict__ b,
                                     unsigned short* __restrict__ hb) {
    long long t = (long long)blockIdx.x * blockDim.x + threadIdx.x;
    if (t >= (long long)NNODES * HC) return;
    float v = h[t] + b[t & (HC - 1)];
    hb[t] = f2bf(v > 0.f ? v : 0.f);
}

// ---------------- layer 2 edge kernels (thread per edge, 10 channels) -------
__global__ void score2_kernel(const int* __restrict__ ei, const float* __restrict__ hl,
                              const float* __restrict__ hr, const float* __restrict__ att,
                              float* __restrict__ sc, unsigned* __restrict__ mkey) {
    int e = blockIdx.x * blockDim.x + threadIdx.x;
    if (e >= EO) return;
    int s, d;
    if (e < NEDGES) { s = ei[e]; d = ei[NEDGES + e]; } else { s = d = e - NEDGES; }
    float acc = 0.f;
    #pragma unroll
    for (int c = 0; c < C2; ++c)
        acc += att[c] * lrelu(hl[(long long)s * C2 + c] + hr[(long long)d * C2 + c]);
    sc[e] = acc;
    atomicMax(&mkey[d], fkey(acc));
}

__global__ void ex2_kernel(const int* __restrict__ ei, float* __restrict__ sc,
                           const unsigned* __restrict__ mkey, float* __restrict__ den) {
    int e = blockIdx.x * blockDim.x + threadIdx.x;
    if (e >= EO) return;
    int d = (e < NEDGES) ? ei[NEDGES + e] : e - NEDGES;
    float ex = __expf(sc[e] - fdec(mkey[d]));
    sc[e] = ex;
    atomicAdd(&den[d], ex);
}

__global__ void agg2_kernel(const int* __restrict__ ei, const float* __restrict__ hl,
                            const float* __restrict__ sc, const float* __restrict__ den,
                            float* __restrict__ out) {
    int e = blockIdx.x * blockDim.x + threadIdx.x;
    if (e >= EO) return;
    int s, d;
    if (e < NEDGES) { s = ei[e]; d = ei[NEDGES + e]; } else { s = d = e - NEDGES; }
    float alpha = sc[e] / den[d];
    #pragma unroll
    for (int c = 0; c < C2; ++c)
        atomicAdd(&out[(long long)d * C2 + c], hl[(long long)s * C2 + c] * alpha);
}

__global__ void bias2_kernel(float* __restrict__ out, const float* __restrict__ b) {
    long long t = (long long)blockIdx.x * blockDim.x + threadIdx.x;
    if (t >= (long long)NNODES * C2) return;
    out[t] += b[t % C2];
}

// ---------------- launch ----------------------------------------------------
extern "C" void kernel_launch(void* const* d_in, const int* in_sizes, int n_in,
                              void* d_out, int out_size, void* d_ws, size_t ws_size,
                              hipStream_t stream) {
    (void)in_sizes; (void)n_in; (void)out_size; (void)ws_size;
    const float* x    = (const float*)d_in[0];
    const int*   ei   = (const int*)d_in[1];      // [2,E]: row0 = src, row1 = dst
    const float* Wl1  = (const float*)d_in[2];
    const float* Wr1  = (const float*)d_in[3];
    const float* att1 = (const float*)d_in[4];
    const float* b1   = (const float*)d_in[5];
    const float* Wl2  = (const float*)d_in[6];
    const float* Wr2  = (const float*)d_in[7];
    const float* att2 = (const float*)d_in[8];
    const float* b2   = (const float*)d_in[9];
    float* out = (float*)d_out;

    float* ws = (float*)d_ws;
    size_t off = 0;   // all block sizes are multiples of 4 floats -> 16B aligned
    float* xl1 = ws + off;                off += (size_t)NNODES * HC;
    float* xr1 = ws + off;                off += (size_t)NNODES * HC;
    float* h1  = ws + off;                off += (size_t)NNODES * HC;
    float* sc1 = ws + off;                off += (size_t)EO * NHEADS;
    unsigned* m1 = (unsigned*)(ws + off); off += (size_t)NNODES * NHEADS;
    float* den1 = ws + off;               off += (size_t)NNODES * NHEADS;
    float* hl2 = ws + off;                off += (size_t)NNODES * C2;
    float* hr2 = ws + off;                off += (size_t)NNODES * C2;
    float* sc2 = ws + off;                off += (size_t)EO;
    unsigned* m2 = (unsigned*)(ws + off); off += (size_t)NNODES;
    float* den2 = ws + off;               off += (size_t)NNODES;
    unsigned short* xb  = (unsigned short*)(ws + off); off += (size_t)NNODES * HC / 2;
    unsigned short* h1b = (unsigned short*)(ws + off); off += (size_t)NNODES * HC / 2;

    auto fill0 = [&](void* p, long long n) {
        fill_zero_kernel<<<(unsigned)((n + 255) / 256), 256, 0, stream>>>((unsigned*)p, n);
    };
    fill0(h1, (long long)NNODES * HC);
    fill0(m1, (long long)NNODES * NHEADS);
    fill0(den1, (long long)NNODES * NHEADS);
    fill0(m2, NNODES);
    fill0(den2, NNODES);
    fill0(out, (long long)NNODES * C2);

    // convert x -> bf16 once (removes all conversion VALU from GEMM hot loop)
    const long long n8 = (long long)NNODES * HC / 8;
    cvt_bf16_kernel<<<(unsigned)((n8 + 255) / 256), 256, 0, stream>>>(x, xb, n8);

    const int Mt = NNODES / 16;          // 3125
    const int Nt1 = HC / 16;             // 16
    const int blocks1 = Nt1 * ((Mt + 3) / 4);
    gemm_wmma_bf16<<<blocks1, 128, 0, stream>>>(xb, Wl1, xl1, Mt, HC, Nt1);
    gemm_wmma_bf16<<<blocks1, 128, 0, stream>>>(xb, Wr1, xr1, Mt, HC, Nt1);

    const int ebW = (EO + 7) / 8;        // 8 waves (edges) per 256-thread block
    score1_kernel<<<ebW, 256, 0, stream>>>(ei, xl1, xr1, att1, sc1, m1);
    const long long t1 = (long long)EO * NHEADS;
    ex1_kernel<<<(unsigned)((t1 + 255) / 256), 256, 0, stream>>>(ei, sc1, m1, den1);
    agg1_kernel<<<ebW, 256, 0, stream>>>(ei, xl1, sc1, den1, h1);
    const long long nh = (long long)NNODES * HC;
    biasrelu_bf16_kernel<<<(unsigned)((nh + 255) / 256), 256, 0, stream>>>(h1, b1, h1b);

    const int blocks2 = (Mt + 3) / 4;    // Ntiles = 1 (10 cols padded to 16)
    gemm_wmma_bf16<<<blocks2, 128, 0, stream>>>(h1b, Wl2, hl2, Mt, C2, 1);
    gemm_wmma_bf16<<<blocks2, 128, 0, stream>>>(h1b, Wr2, hr2, Mt, C2, 1);

    const int eb = (EO + 255) / 256;
    score2_kernel<<<eb, 256, 0, stream>>>(ei, hl2, hr2, att2, sc2, m2);
    ex2_kernel<<<eb, 256, 0, stream>>>(ei, sc2, m2, den2);
    agg2_kernel<<<eb, 256, 0, stream>>>(ei, hl2, sc2, den2, out);
    const long long no = (long long)NNODES * C2;
    bias2_kernel<<<(unsigned)((no + 255) / 256), 256, 0, stream>>>(out, b2);
}
